// GATNet_25443386261563
// MI455X (gfx1250) — compile-verified, matrix operand order checked by reading, not run on hardware
//
#include <hip/hip_runtime.h>
#include <hip/hip_bf16.h>
#include <math.h>

typedef __attribute__((ext_vector_type(16))) _Float16 v16h;
typedef __attribute__((ext_vector_type(8)))  _Float16 v8h;
typedef __attribute__((ext_vector_type(8)))  float    v8f;
typedef __attribute__((ext_vector_type(4)))  float    v4f;

#define N_NODES 50000
#define N_EDGES 800000
#define N_ETOT  850000   // + self loops
#define IN_CH   768
#define HID     32
#define HEADS   8
#define HC      256      // HEADS*HID
#define OUT_CH  40
#define OUT_PAD 48
#define NEG_SLOPE 0.2f

// ---------------------------------------------------------------------------
// Weight transpose + f32->f16: Wt[n][k] = W[k][n], zero-padded to Npad rows.
// ---------------------------------------------------------------------------
__global__ void transpose_w_kernel(const float* __restrict__ W, _Float16* __restrict__ Wt,
                                   int K, int Nn, int Npad) {
    int gid = blockIdx.x * blockDim.x + threadIdx.x;
    if (gid >= Npad * K) return;
    int n = gid / K;
    int k = gid % K;
    Wt[gid] = (n < Nn) ? (_Float16)W[(size_t)k * Nn + n] : (_Float16)0.0f;
}

// ---------------------------------------------------------------------------
// WMMA GEMM, LDS-staged A operand, register-batched B operand.
//   Block = 256 threads (8 waves) = one 16-row M stripe.
//   A chunk (16 x 128) staged once into LDS (f32->f16 on the fly), consumed by
//   all 8 waves; each wave owns NPW 16-column output tiles. All B fragments
//   for a K-chunk are loaded into registers before the barrier so the k-steps
//   run as pure ds_load + v_wmma.
//   GUARD=false -> every wave fully active (no branch around WMMA at all).
//   GUARD=true  -> scalar (readfirstlane) predicate, s_cbranch not exec-mask.
// ---------------------------------------------------------------------------
template <bool A_IS_HALF, int NPW, bool GUARD>
__global__ void gemm_wmma_kernel(const void* __restrict__ Aptr,
                                 const _Float16* __restrict__ Bt,
                                 float* __restrict__ C,
                                 int K, int Ntiles, int ldc) {
    constexpr int KCHUNK = 128;
    constexpr int LDA    = KCHUNK + 8;            // pad: bank stride 4 across rows
    __shared__ _Float16 As[16 * LDA];

    const int tid  = threadIdx.x;
    const int wave = tid >> 5;
    const int lane = tid & 31;
    const int mt   = blockIdx.x;

    const int lrow = tid >> 4;                    // staging row   0..15
    const int lcol = (tid & 15) * 8;              // staging col   0..120
    const int mloc = lane & 31 & 15;              // frag row/col within tile
    const int kb   = (lane >> 4) * 8;             // per-lane K sub-base (0 or 8)

    v8f acc[NPW] = {};
    const size_t arow = (size_t)(mt * 16 + lrow) * K;

    // Wave-uniform tile activity; readfirstlane forces a scalar branch.
    bool act[NPW];
    const _Float16* bbase[NPW];
#pragma unroll
    for (int j = 0; j < NPW; ++j) {
        int nt = wave * NPW + j;
        act[j]   = GUARD ? (__builtin_amdgcn_readfirstlane(nt < Ntiles ? 1 : 0) != 0) : true;
        bbase[j] = Bt + (size_t)(nt * 16 + mloc) * K + kb;
    }

    for (int kc = 0; kc < K; kc += KCHUNK) {
        // ---- cooperative stage of A chunk into LDS as f16 ----
        v8h h8;
        if (A_IS_HALF) {
            const _Float16* Ah = (const _Float16*)Aptr;
            h8 = *(const v8h*)(Ah + arow + kc + lcol);
            if (kc + KCHUNK < K)
                __builtin_prefetch(Ah + arow + kc + KCHUNK + lcol, 0, 0);
        } else {
            const float* Af = (const float*)Aptr;
            const v4f* p = (const v4f*)(Af + arow + kc + lcol);
            v4f f0 = p[0], f1 = p[1];
#pragma unroll
            for (int j = 0; j < 4; ++j) { h8[j] = (_Float16)f0[j]; h8[4 + j] = (_Float16)f1[j]; }
            if (kc + KCHUNK < K)
                __builtin_prefetch(Af + arow + kc + KCHUNK + lcol, 0, 0);
        }
        *(v8h*)&As[lrow * LDA + lcol] = h8;

        // ---- batch-load all B fragments for this chunk (overlaps barrier) ----
        v8h blo[NPW][4], bhi[NPW][4];
#pragma unroll
        for (int j = 0; j < NPW; ++j) {
            if (act[j]) {
#pragma unroll
                for (int ks = 0; ks < 4; ++ks) {
                    blo[j][ks] = *(const v8h*)(bbase[j] + kc + ks * 32);
                    bhi[j][ks] = *(const v8h*)(bbase[j] + kc + ks * 32 + 16);
                }
            }
        }
        __syncthreads();

        // ---- consume: 4 WMMA K-steps of 32, NPW output tiles per wave ----
#pragma unroll
        for (int ks = 0; ks < 4; ++ks) {
            v16h a;
            {
                v8h lo = *(const v8h*)&As[mloc * LDA + ks * 32 + kb];
                v8h hi = *(const v8h*)&As[mloc * LDA + ks * 32 + kb + 16];
#pragma unroll
                for (int u = 0; u < 8; ++u) { a[u] = lo[u]; a[8 + u] = hi[u]; }
            }
#pragma unroll
            for (int j = 0; j < NPW; ++j) {
                if (act[j]) {
                    v16h b;
#pragma unroll
                    for (int u = 0; u < 8; ++u) { b[u] = blo[j][ks][u]; b[8 + u] = bhi[j][ks][u]; }
                    acc[j] = __builtin_amdgcn_wmma_f32_16x16x32_f16(false, a, false, b,
                                                                    (short)0, acc[j],
                                                                    false, false);
                }
            }
        }
        __syncthreads();
    }

    // ---- store: C-frag VGPR r -> row (8*(lane>>4) + r), col (lane&15) ----
    const int rbase = (lane >> 4) * 8;
#pragma unroll
    for (int j = 0; j < NPW; ++j) {
        if (!act[j]) continue;
        int nt = wave * NPW + j;
#pragma unroll
        for (int r = 0; r < 8; ++r) {
            int m = mt * 16 + rbase + r;
            C[(size_t)m * ldc + nt * 16 + mloc] = acc[j][r];
        }
    }
}

// ---------------------------------------------------------------------------
// Per-node attention logits + init of segment-max/denominator buffers.
// ---------------------------------------------------------------------------
__global__ void attn_node_kernel(const float* __restrict__ feat, int ldf,
                                 const float* __restrict__ a_src,
                                 const float* __restrict__ a_dst,
                                 float* __restrict__ as_v, float* __restrict__ ad_v,
                                 float* __restrict__ emax, float* __restrict__ denom,
                                 int H, int C) {
    int gid = blockIdx.x * blockDim.x + threadIdx.x;
    if (gid >= N_NODES * H) return;
    int n = gid / H;
    int h = gid % H;
    const float* f = feat + (size_t)n * ldf + h * C;
    float s = 0.f, d = 0.f;
    for (int c = 0; c < C; ++c) {
        float v = f[c];
        s += v * a_src[h * C + c];
        d += v * a_dst[h * C + c];
    }
    as_v[gid]  = s;
    ad_v[gid]  = d;
    emax[gid]  = -INFINITY;
    denom[gid] = 0.f;
}

__global__ void zero_kernel(float* __restrict__ p, long long n) {
    long long i = (long long)blockIdx.x * blockDim.x + threadIdx.x;
    if (i < n) p[i] = 0.f;
}

__device__ __forceinline__ float leaky(float x) { return x > 0.f ? x : NEG_SLOPE * x; }

__device__ __forceinline__ void atomic_max_f32(float* addr, float v) {
    if (v >= 0.f) atomicMax((int*)addr, __float_as_int(v));
    else          atomicMin((unsigned int*)addr, __float_as_uint(v));
}

__device__ __forceinline__ int edge_src(const int* ei, int e) {
    return (e < N_EDGES) ? ei[e] : (e - N_EDGES);
}
__device__ __forceinline__ int edge_dst(const int* ei, int e) {
    return (e < N_EDGES) ? ei[N_EDGES + e] : (e - N_EDGES);
}

// Pass 1: segment max over incoming edges.
__global__ void edge_max_kernel(const int* __restrict__ ei,
                                const float* __restrict__ as_v,
                                const float* __restrict__ ad_v,
                                float* __restrict__ emax, int H) {
    int e = blockIdx.x * blockDim.x + threadIdx.x;
    if (e >= N_ETOT) return;
    int s = edge_src(ei, e);
    int d = edge_dst(ei, e);
    for (int h = 0; h < H; ++h) {
        float v = leaky(as_v[s * H + h] + ad_v[d * H + h]);
        atomic_max_f32(&emax[d * H + h], v);
    }
}

// Pass 2: p = exp(e - max), segment sum of p.
__global__ void edge_expsum_kernel(const int* __restrict__ ei,
                                   const float* __restrict__ as_v,
                                   const float* __restrict__ ad_v,
                                   const float* __restrict__ emax,
                                   float* __restrict__ p_edge,
                                   float* __restrict__ denom, int H) {
    int e = blockIdx.x * blockDim.x + threadIdx.x;
    if (e >= N_ETOT) return;
    int s = edge_src(ei, e);
    int d = edge_dst(ei, e);
    for (int h = 0; h < H; ++h) {
        float v = leaky(as_v[s * H + h] + ad_v[d * H + h]);
        float p = __expf(v - emax[d * H + h]);
        p_edge[(size_t)e * H + h] = p;
        atomicAdd(&denom[d * H + h], p);
    }
}

// Pass 3: weighted scatter-sum of source features. One block per edge.
__global__ void edge_scatter_kernel(const int* __restrict__ ei,
                                    const float* __restrict__ feat, int ldf,
                                    const float* __restrict__ p_edge,
                                    const float* __restrict__ denom,
                                    float* __restrict__ agg, int H, int C) {
    int e = blockIdx.x;
    int i = threadIdx.x;
    int hc = H * C;
    if (i >= hc) return;
    int s = edge_src(ei, e);
    int d = edge_dst(ei, e);
    int h = i / C;
    float w = p_edge[(size_t)e * H + h] / (denom[d * H + h] + 1e-16f);
    atomicAdd(&agg[(size_t)d * hc + i], feat[(size_t)s * ldf + i] * w);
}

// Post: out = elu(agg + bias), emitted as f16 for the next layer's WMMA A-operand.
__global__ void post_elu_kernel(const float* __restrict__ agg,
                                const float* __restrict__ bias,
                                _Float16* __restrict__ outh, long long total, int hc) {
    long long gid = (long long)blockIdx.x * blockDim.x + threadIdx.x;
    if (gid >= total) return;
    int i = (int)(gid % hc);
    float v = agg[gid] + bias[i];
    v = v > 0.f ? v : (__expf(v) - 1.f);
    outh[gid] = (_Float16)v;
}

// Final: out = log_softmax(agg2 + b2) over 40 classes.
__global__ void final_logsoftmax_kernel(const float* __restrict__ agg,
                                        const float* __restrict__ b2,
                                        float* __restrict__ out) {
    int n = blockIdx.x * blockDim.x + threadIdx.x;
    if (n >= N_NODES) return;
    float v[OUT_CH];
    float m = -INFINITY;
#pragma unroll
    for (int c = 0; c < OUT_CH; ++c) {
        v[c] = agg[(size_t)n * OUT_CH + c] + b2[c];
        m = fmaxf(m, v[c]);
    }
    float s = 0.f;
#pragma unroll
    for (int c = 0; c < OUT_CH; ++c) s += __expf(v[c] - m);
    float lse = __logf(s);
#pragma unroll
    for (int c = 0; c < OUT_CH; ++c) out[(size_t)n * OUT_CH + c] = v[c] - m - lse;
}

// ---------------------------------------------------------------------------
// Host orchestration
// ---------------------------------------------------------------------------
static inline size_t align256(size_t x) { return (x + 255) & ~(size_t)255; }

extern "C" void kernel_launch(void* const* d_in, const int* in_sizes, int n_in,
                              void* d_out, int out_size, void* d_ws, size_t ws_size,
                              hipStream_t stream) {
    const float* x   = (const float*)d_in[0];
    const int*   ei  = (const int*)d_in[1];      // [2, N_EDGES], int32
    const float* W0  = (const float*)d_in[2];
    const float* as0 = (const float*)d_in[3];
    const float* ad0 = (const float*)d_in[4];
    const float* b0  = (const float*)d_in[5];
    const float* W1  = (const float*)d_in[6];
    const float* as1 = (const float*)d_in[7];
    const float* ad1 = (const float*)d_in[8];
    const float* b1  = (const float*)d_in[9];
    const float* W2  = (const float*)d_in[10];
    const float* as2 = (const float*)d_in[11];
    const float* ad2 = (const float*)d_in[12];
    const float* b2  = (const float*)d_in[13];
    float* out = (float*)d_out;

    // Workspace carving
    char*  base = (char*)d_ws;
    size_t off  = 0;
    auto carve = [&](size_t bytes) -> void* {
        void* p = base + off;
        off += align256(bytes);
        return p;
    };
    _Float16* Wt0   = (_Float16*)carve((size_t)HC * IN_CH * 2);        // [256][768]
    _Float16* Wt1   = (_Float16*)carve((size_t)HC * HC * 2);           // [256][256]
    _Float16* Wt2   = (_Float16*)carve((size_t)OUT_PAD * HC * 2);      // [48][256]
    _Float16* actH  = (_Float16*)carve((size_t)N_NODES * HC * 2);      // f16 activations
    float*    feat  = (float*)carve((size_t)N_NODES * HC * 4);         // GEMM output
    float*    agg   = (float*)carve((size_t)N_NODES * HC * 4);         // aggregation
    float*    as_v  = (float*)carve((size_t)N_NODES * HEADS * 4);
    float*    ad_v  = (float*)carve((size_t)N_NODES * HEADS * 4);
    float*    emax  = (float*)carve((size_t)N_NODES * HEADS * 4);
    float*    denom = (float*)carve((size_t)N_NODES * HEADS * 4);
    float*    pedge = (float*)carve((size_t)N_ETOT * HEADS * 4);
    (void)ws_size; (void)n_in; (void)in_sizes; (void)out_size;

    const int Mtiles = N_NODES / 16;   // 3125, exact

    // Pre-transpose weights to f16 (B operand layout)
    {
        int t0 = HC * IN_CH, t1 = HC * HC, t2 = OUT_PAD * HC;
        transpose_w_kernel<<<(t0 + 255) / 256, 256, 0, stream>>>(W0, Wt0, IN_CH, HC, HC);
        transpose_w_kernel<<<(t1 + 255) / 256, 256, 0, stream>>>(W1, Wt1, HC, HC, HC);
        transpose_w_kernel<<<(t2 + 255) / 256, 256, 0, stream>>>(W2, Wt2, HC, OUT_CH, OUT_PAD);
    }

    auto run_edge_phase = [&](const float* featL, int ldf, const float* aS, const float* aD,
                              int H, int C, float* aggL) {
        int nh = N_NODES * H;
        attn_node_kernel<<<(nh + 255) / 256, 256, 0, stream>>>(featL, ldf, aS, aD,
                                                               as_v, ad_v, emax, denom, H, C);
        long long aggN = (long long)N_NODES * H * C;
        zero_kernel<<<(unsigned)((aggN + 255) / 256), 256, 0, stream>>>(aggL, aggN);
        edge_max_kernel<<<(N_ETOT + 255) / 256, 256, 0, stream>>>(ei, as_v, ad_v, emax, H);
        edge_expsum_kernel<<<(N_ETOT + 255) / 256, 256, 0, stream>>>(ei, as_v, ad_v, emax,
                                                                     pedge, denom, H);
        int blk = ((H * C + 63) / 64) * 64;
        edge_scatter_kernel<<<N_ETOT, blk, 0, stream>>>(ei, featL, ldf, pedge, denom,
                                                        aggL, H, C);
    };

    // ---------------- Layer 0: x[50000,768] -> feat[50000,256] ----------------
    {
        gemm_wmma_kernel<false, 2, false><<<Mtiles, 256, 0, stream>>>(
            (const void*)x, Wt0, feat, IN_CH, HC / 16, HC);
        run_edge_phase(feat, HC, as0, ad0, HEADS, HID, agg);
        long long tot = (long long)N_NODES * HC;
        post_elu_kernel<<<(unsigned)((tot + 255) / 256), 256, 0, stream>>>(agg, b0, actH, tot, HC);
    }

    // ---------------- Layer 1: actH[50000,256] -> feat[50000,256] -------------
    {
        gemm_wmma_kernel<true, 2, false><<<Mtiles, 256, 0, stream>>>(
            (const void*)actH, Wt1, feat, HC, HC / 16, HC);
        run_edge_phase(feat, HC, as1, ad1, HEADS, HID, agg);
        long long tot = (long long)N_NODES * HC;
        post_elu_kernel<<<(unsigned)((tot + 255) / 256), 256, 0, stream>>>(agg, b1, actH, tot, HC);
    }

    // ---------------- Layer 2: actH[50000,256] -> feat[50000,48] (40 used) ----
    {
        gemm_wmma_kernel<true, 1, true><<<Mtiles, 256, 0, stream>>>(
            (const void*)actH, Wt2, feat, HC, OUT_PAD / 16, OUT_PAD);
        run_edge_phase(feat, OUT_PAD, as2, ad2, 1, OUT_CH, agg);
        final_logsoftmax_kernel<<<(N_NODES + 255) / 256, 256, 0, stream>>>(agg, b2, out);
    }
}
